// VLALayer_33775622816412
// MI455X (gfx1250) — compile-verified, matrix operand order checked by reading, not run on hardware
//
#include <hip/hip_runtime.h>
#include <hip/hip_bf16.h>

// ---------------------------------------------------------------------------
// VLA layer for MI455X (gfx1250, wave32):
//   phase 1: QKV projection GEMMs  -> v_wmma_f32_16x16x32_bf16
//   phase 2: Sherman-Morrison scan -> VALU + LDS, async-to-LDS double buffering
//   phase 3: output projection GEMM-> v_wmma_f32_16x16x32_bf16, NT stores
// ---------------------------------------------------------------------------

typedef __attribute__((ext_vector_type(16))) __bf16 v16bf;
typedef __attribute__((ext_vector_type(8)))  float  v8f;

#define T_SEQ   2048
#define DMODEL  512
#define DHEAD   64
#define BATCH   16
#define ROWS    (BATCH * T_SEQ)   // 32768

// Pack 8 consecutive f32 (16B-aligned) into bf16 fragment elements [base..base+7].
__device__ __forceinline__ void pack8(const float* __restrict__ p, v16bf& dst, int base) {
    const float4 f0 = *reinterpret_cast<const float4*>(p);
    const float4 f1 = *reinterpret_cast<const float4*>(p + 4);
    dst[base + 0] = (__bf16)f0.x; dst[base + 1] = (__bf16)f0.y;
    dst[base + 2] = (__bf16)f0.z; dst[base + 3] = (__bf16)f0.w;
    dst[base + 4] = (__bf16)f1.x; dst[base + 5] = (__bf16)f1.y;
    dst[base + 6] = (__bf16)f1.z; dst[base + 7] = (__bf16)f1.w;
}

// ---------------------------------------------------------------------------
// Phase 1: q/k/v = x @ W^T + b.  One wave computes a 16x64 tile.
// A fragment (16x32 bf16): lane = M row (lane&15), K-half by lane>>4.
// B fragment (32x16 bf16): lane = N col, same K packing; W rows are K-contig.
// ---------------------------------------------------------------------------
__global__ __launch_bounds__(32) void vla_qkv_proj(
    const float* __restrict__ x,
    const float* __restrict__ Wq, const float* __restrict__ bq,
    const float* __restrict__ Wk, const float* __restrict__ bk,
    const float* __restrict__ Wv, const float* __restrict__ bv,
    float* __restrict__ Qo, float* __restrict__ Ko, float* __restrict__ Vo)
{
    const int lane  = threadIdx.x;
    const int row0  = blockIdx.x * 16;
    const int which = blockIdx.y;

    const float* W    = (which == 0) ? Wq : (which == 1) ? Wk : Wv;
    const float* bias = (which == 0) ? bq : (which == 1) ? bk : bv;
    float*       out  = (which == 0) ? Qo : (which == 1) ? Ko : Vo;

    const int mr    = lane & 15;
    const int khalf = (lane >> 4) * 8;

    v8f acc[4];
    #pragma unroll
    for (int n = 0; n < 4; ++n)
        #pragma unroll
        for (int i = 0; i < 8; ++i) acc[n][i] = 0.0f;

    const float* xrow = x + (size_t)(row0 + mr) * DMODEL;

    for (int k0 = 0; k0 < DMODEL; k0 += 32) {
        __builtin_prefetch(xrow + k0 + 128, 0, 1);   // global_prefetch_b8
        v16bf a;
        pack8(xrow + k0 + khalf,      a, 0);
        pack8(xrow + k0 + 16 + khalf, a, 8);
        #pragma unroll
        for (int n = 0; n < 4; ++n) {
            const float* wrow = W + (size_t)(n * 16 + mr) * DMODEL + k0;
            v16bf bfrag;
            pack8(wrow + khalf,      bfrag, 0);
            pack8(wrow + 16 + khalf, bfrag, 8);
            acc[n] = __builtin_amdgcn_wmma_f32_16x16x32_bf16(
                false, a, false, bfrag, (short)0, acc[n], false, false);
        }
    }

    // C/D layout: VGPR i -> M = i + 8*(lane>>4), N = lane&15
    const int mbase = (lane >> 4) * 8;
    #pragma unroll
    for (int n = 0; n < 4; ++n) {
        const float bz = bias[n * 16 + mr];
        #pragma unroll
        for (int i = 0; i < 8; ++i)
            out[(size_t)(row0 + mbase + i) * DHEAD + n * 16 + mr] = acc[n][i] + bz;
    }
}

// ---------------------------------------------------------------------------
// Phase 2: sequential Sherman-Morrison scan, one batch per block.
// 256 threads: thread (r = tid>>2, cq = tid&3) owns A[r][cq*16..+16) and
// S[r][cq*16..+16) in registers; the 4 slices of a row live in adjacent
// lanes so row reductions are two intra-wave shuffles.
// k/q/v for step t+1 are staged into LDS with global_load_async_to_lds_b32
// (ASYNCcnt) while step t computes; z_t = A_old u / denom simplification.
// ---------------------------------------------------------------------------
__global__ __launch_bounds__(256) void vla_scan(
    const float* __restrict__ Q, const float* __restrict__ K,
    const float* __restrict__ V, float* __restrict__ O)
{
    __shared__ float kqv[2][192];   // [buf][0:64)=k, [64:128)=q, [128:192)=v
    __shared__ float Au_s[64];
    __shared__ float red[8];

    const int tid = threadIdx.x;
    const int r   = tid >> 2;
    const int cq  = tid & 3;
    const int j0  = cq * 16;
    const int b   = blockIdx.x;

    float A[16], S[16];
    #pragma unroll
    for (int j = 0; j < 16; ++j) {
        A[j] = (r == j0 + j) ? 1.0f : 0.0f;   // A_0 = I / lambda_0, lambda_0 = 1
        S[j] = 0.0f;
    }

    const size_t base = (size_t)b * T_SEQ * DHEAD;
    const float* src  = (tid < 64) ? K : (tid < 128) ? Q : V;
    const int    d    = tid & 63;

    // prologue: async-stage step 0 into buffer 0
    if (tid < 192) {
        const unsigned long long g =
            (unsigned long long)(uintptr_t)(src + base + d);
        const unsigned lds = (unsigned)(uintptr_t)&kqv[0][tid];
        asm volatile("global_load_async_to_lds_b32 %0, %1, off"
                     :: "v"(lds), "v"(g) : "memory");
    }
    asm volatile("s_wait_asynccnt 0x0" ::: "memory");
    __syncthreads();

    for (int t = 0; t < T_SEQ; ++t) {
        const int cur = t & 1;
        const float* kc = &kqv[cur][0];
        const float* qc = &kqv[cur][64];
        const float* vc = &kqv[cur][128];

        // kick off staging of step t+1 into the other buffer (hidden latency)
        if ((t + 1 < T_SEQ) && tid < 192) {
            const unsigned long long g =
                (unsigned long long)(uintptr_t)(src + base + (size_t)(t + 1) * DHEAD + d);
            const unsigned lds = (unsigned)(uintptr_t)&kqv[cur ^ 1][tid];
            asm volatile("global_load_async_to_lds_b32 %0, %1, off"
                         :: "v"(lds), "v"(g) : "memory");
        }

        // Au[r] = A[r][:] . k  (16-wide partial + 2 intra-wave shuffles)
        float p = 0.0f;
        #pragma unroll
        for (int j = 0; j < 16; ++j) p = fmaf(A[j], kc[j0 + j], p);
        p += __shfl_xor(p, 1, 32);
        p += __shfl_xor(p, 2, 32);
        const float Au_r = p;
        if (cq == 0) Au_s[r] = Au_r;
        __syncthreads();

        // denom = 1 + u.Au ; sdot = k.q  (4 threads, 16-wide partials)
        if (tid < 4) {
            const int jb = tid * 16;
            float pd = 0.0f, ps = 0.0f;
            #pragma unroll
            for (int j = 0; j < 16; ++j) {
                const float kk = kc[jb + j];
                pd = fmaf(kk, Au_s[jb + j], pd);
                ps = fmaf(kk, qc[jb + j], ps);
            }
            red[tid]     = pd;
            red[4 + tid] = ps;
        }
        __syncthreads();

        const float denom = 1.0f + red[0] + red[1] + red[2] + red[3];
        const float sdot  = red[4] + red[5] + red[6] + red[7];
        const float inv   = 1.0f / denom;

        // A -= (Au Au^T)/denom ; alpha = sdot*inv*Au ; S += v alpha^T ;
        // o partial = S[r][:] . q  — fused in one slice loop.
        const float ca = -Au_r * inv;
        const float cs = vc[r] * sdot * inv;
        float po = 0.0f;
        #pragma unroll
        for (int j = 0; j < 16; ++j) {
            const float auj = Au_s[j0 + j];
            A[j] = fmaf(ca, auj, A[j]);
            S[j] = fmaf(cs, auj, S[j]);
            po   = fmaf(S[j], qc[j0 + j], po);
        }
        po += __shfl_xor(po, 1, 32);
        po += __shfl_xor(po, 2, 32);
        if (cq == 0) O[base + (size_t)t * DHEAD + r] = po;

        // next-step staging must be complete before buffers flip
        asm volatile("s_wait_asynccnt 0x0" ::: "memory");
        __syncthreads();
    }
}

// ---------------------------------------------------------------------------
// Phase 3: y = O @ W_o^T + b_o.  M=32768, N=512, K=64.  16x64 tile per wave.
// y is write-once (64 MB) -> non-temporal stores to keep it out of L2.
// ---------------------------------------------------------------------------
__global__ __launch_bounds__(32) void vla_out_proj(
    const float* __restrict__ Oin, const float* __restrict__ Wo,
    const float* __restrict__ bo, float* __restrict__ y)
{
    const int lane  = threadIdx.x;
    const int row0  = blockIdx.x * 16;
    const int nbase = blockIdx.y * 64;
    const int mr    = lane & 15;
    const int khalf = (lane >> 4) * 8;

    v8f acc[4];
    #pragma unroll
    for (int n = 0; n < 4; ++n)
        #pragma unroll
        for (int i = 0; i < 8; ++i) acc[n][i] = 0.0f;

    const float* arow = Oin + (size_t)(row0 + mr) * DHEAD;

    #pragma unroll
    for (int k0 = 0; k0 < DHEAD; k0 += 32) {
        v16bf a;
        pack8(arow + k0 + khalf,      a, 0);
        pack8(arow + k0 + 16 + khalf, a, 8);
        #pragma unroll
        for (int n = 0; n < 4; ++n) {
            const float* wrow = Wo + (size_t)(nbase + n * 16 + mr) * DHEAD + k0;
            v16bf bfrag;
            pack8(wrow + khalf,      bfrag, 0);
            pack8(wrow + 16 + khalf, bfrag, 8);
            acc[n] = __builtin_amdgcn_wmma_f32_16x16x32_bf16(
                false, a, false, bfrag, (short)0, acc[n], false, false);
        }
    }

    const int mbase = (lane >> 4) * 8;
    #pragma unroll
    for (int n = 0; n < 4; ++n) {
        const float bz = bo[nbase + n * 16 + mr];
        #pragma unroll
        for (int i = 0; i < 8; ++i)
            __builtin_nontemporal_store(
                acc[n][i] + bz,
                &y[(size_t)(row0 + mbase + i) * DMODEL + nbase + n * 16 + mr]);
    }
}

// ---------------------------------------------------------------------------
extern "C" void kernel_launch(void* const* d_in, const int* in_sizes, int n_in,
                              void* d_out, int out_size, void* d_ws, size_t ws_size,
                              hipStream_t stream) {
    const float* x  = (const float*)d_in[0];
    const float* Wq = (const float*)d_in[1];
    const float* bq = (const float*)d_in[2];
    const float* Wk = (const float*)d_in[3];
    const float* bk = (const float*)d_in[4];
    const float* Wv = (const float*)d_in[5];
    const float* bv = (const float*)d_in[6];
    const float* Wo = (const float*)d_in[7];
    const float* bo = (const float*)d_in[8];
    float* y = (float*)d_out;

    // workspace: Q, K, V, O intermediates (each 32768 x 64 f32 = 8 MB)
    float* Qb = (float*)d_ws;
    float* Kb = Qb + (size_t)ROWS * DHEAD;
    float* Vb = Kb + (size_t)ROWS * DHEAD;
    float* Ob = Vb + (size_t)ROWS * DHEAD;

    vla_qkv_proj<<<dim3(ROWS / 16, 3), 32, 0, stream>>>(
        x, Wq, bq, Wk, bk, Wv, bv, Qb, Kb, Vb);

    vla_scan<<<dim3(BATCH), 256, 0, stream>>>(Qb, Kb, Vb, Ob);

    vla_out_proj<<<dim3(ROWS / 16, DMODEL / 64), 32, 0, stream>>>(Ob, Wo, bo, y);
}
